// SparseAttention_41317585387918
// MI455X (gfx1250) — matrix operand order repaired and checked by reading, not compiled, not run on hardware
//
#include <hip/hip_runtime.h>

typedef __attribute__((ext_vector_type(16))) _Float16 v16h;
typedef __attribute__((ext_vector_type(8)))  float    v8f;
typedef __attribute__((ext_vector_type(4)))  float    v4f;

#define HD    64
#define SEQ   2048
#define QSCL  0.125f   /* 1/sqrt(64) */

// A-operand fragment (f16 16x32): lane&15 = M row; lanes<16 hold K = c*32+0..15,
// lanes>=16 hold K = c*32+16..31.  rowp points at the f32 row (length 64).
__device__ __forceinline__ v16h load_fragA(const float* rowp, int c, int hi8) {
  const float* p = rowp + c * 32 + hi8;
  v16h f;
#pragma unroll
  for (int i = 0; i < 8; ++i) f[i] = (_Float16)p[i];
#pragma unroll
  for (int i = 0; i < 8; ++i) f[8 + i] = (_Float16)p[8 + i];
  return f;
}

// B-operand fragment (f16 32x16): lane&15 = N col; lanes<16 hold K = c*32+0..15,
// lanes>=16 hold K = c*32+16..31.  rowp points at the f32 row (row-major, we feed Q^T).
__device__ __forceinline__ v16h load_fragB_row(const float* rowp, int c, int hi16) {
  const float* p = rowp + c * 32 + hi16;
  v16h f;
#pragma unroll
  for (int i = 0; i < 16; ++i) f[i] = (_Float16)p[i];
  return f;
}

// S^T tile (16k x 16q) for k-tile starting at k0:  S^T = K_tile * Q_tile^T  (K-dim = d = 64)
__device__ __forceinline__ v8f st_tile(const float* Kb, int k0, v16h qb0, v16h qb1, int lane) {
  const int hi8 = (lane & 16) ? 16 : 0;
  const float* rowp = Kb + (size_t)(k0 + (lane & 15)) * HD;
  v16h ka0 = load_fragA(rowp, 0, hi8);
  v16h ka1 = load_fragA(rowp, 1, hi8);
  v8f acc = {0.f, 0.f, 0.f, 0.f, 0.f, 0.f, 0.f, 0.f};
  acc = __builtin_amdgcn_wmma_f32_16x16x32_f16(false, ka0, false, qb0, (short)0, acc, false, false);
  acc = __builtin_amdgcn_wmma_f32_16x16x32_f16(false, ka1, false, qb1, (short)0, acc, false, false);
  return acc;
}

__global__ __launch_bounds__(256, 1)
void attn_fwd_causal(const float* __restrict__ Q, const float* __restrict__ Km,
                     const float* __restrict__ Vm, float* __restrict__ Out,
                     float* __restrict__ W) {
  const int lane = threadIdx.x & 31;
  const int wid  = blockIdx.x * (blockDim.x >> 5) + (threadIdx.x >> 5);
  const int bh   = wid >> 7;    // 0..31  (batch*head)
  const int qt   = wid & 127;   // q tile index (16 rows each)
  const int q0   = qt * 16;

  const float* Qb = Q  + (size_t)bh * SEQ * HD;
  const float* Kb = Km + (size_t)bh * SEQ * HD;
  const float* Vb = Vm + (size_t)bh * SEQ * HD;
  float* Ob = Out + (size_t)bh * SEQ * HD;
  float* Wb = W   + (size_t)bh * SEQ * SEQ;

  const int hi   = (lane >> 4) & 1;
  const int hi8  = hi * 8;
  const int hi16 = hi * 16;
  const int qrow = q0 + (lane & 15);

  // Q^T as B operand, kept live for the whole kernel (2 x v16h)
  const float* qrp = Qb + (size_t)qrow * HD;
  v16h qb0 = load_fragB_row(qrp, 0, hi16);
  v16h qb1 = load_fragB_row(qrp, 1, hi16);

  // ---------------- pass 1: online row max + exp-sum ----------------
  float m = -__builtin_inff();
  float l = 0.f;
  for (int kt = 0; kt <= qt; ++kt) {
    v8f s = st_tile(Kb, kt * 16, qb0, qb1, lane);
    float sv[8];
    float mx = -__builtin_inff();
#pragma unroll
    for (int v = 0; v < 8; ++v) {
      const int kk = kt * 16 + v + hi8;           // k row of this element
      float x = s[v] * QSCL;
      x = (kk > qrow) ? -__builtin_inff() : x;    // causal mask
      sv[v] = x;
      mx = fmaxf(mx, x);
    }
    mx = fmaxf(mx, __shfl_xor(mx, 16, 32));       // combine lane-halves (same q row)
    const float mn = fmaxf(m, mx);
    float sum = 0.f;
#pragma unroll
    for (int v = 0; v < 8; ++v) sum += __expf(sv[v] - mn);
    sum += __shfl_xor(sum, 16, 32);
    l = l * __expf(m - mn) + sum;
    m = mn;
  }
  const float inv_l = 1.0f / l;

  // ---------------- pass 2: weights + O accumulation ----------------
  v8f oacc[4];
#pragma unroll
  for (int n = 0; n < 4; ++n) oacc[n] = (v8f){0.f, 0.f, 0.f, 0.f, 0.f, 0.f, 0.f, 0.f};

  for (int kt = 0; kt <= qt; kt += 2) {
    float p0[8], p1[8];
    {
      v8f s = st_tile(Kb, kt * 16, qb0, qb1, lane);
#pragma unroll
      for (int v = 0; v < 8; ++v) {
        const int kk = kt * 16 + v + hi8;
        float x = s[v] * QSCL;
        x = (kk > qrow) ? -__builtin_inff() : x;
        p0[v] = __expf(x - m) * inv_l;            // masked -> exp(-inf) = 0
      }
      // streaming weights: non-temporal (TH=NT) so the 537MB stream doesn't evict K/V from L2
      v4f* wp = (v4f*)(Wb + (size_t)qrow * SEQ + kt * 16 + hi8);
      __builtin_nontemporal_store((v4f){p0[0], p0[1], p0[2], p0[3]}, wp + 0);
      __builtin_nontemporal_store((v4f){p0[4], p0[5], p0[6], p0[7]}, wp + 1);
    }
    if (kt + 1 <= qt) {
      v8f s = st_tile(Kb, (kt + 1) * 16, qb0, qb1, lane);
#pragma unroll
      for (int v = 0; v < 8; ++v) {
        const int kk = (kt + 1) * 16 + v + hi8;
        float x = s[v] * QSCL;
        x = (kk > qrow) ? -__builtin_inff() : x;
        p1[v] = __expf(x - m) * inv_l;
      }
      v4f* wp = (v4f*)(Wb + (size_t)qrow * SEQ + (kt + 1) * 16 + hi8);
      __builtin_nontemporal_store((v4f){p1[0], p1[1], p1[2], p1[3]}, wp + 0);
      __builtin_nontemporal_store((v4f){p1[4], p1[5], p1[6], p1[7]}, wp + 1);
    } else {
#pragma unroll
      for (int v = 0; v < 8; ++v) p1[v] = 0.f;
    }

    // P (in S^T C-layout) IS the A-operand layout for O = P*V: just pack f32->f16.
    v16h pa;
#pragma unroll
    for (int v = 0; v < 8; ++v) pa[v] = (_Float16)p0[v];
#pragma unroll
    for (int v = 0; v < 8; ++v) pa[8 + v] = (_Float16)p1[v];

    const int kb = kt * 16;
#pragma unroll
    for (int n = 0; n < 4; ++n) {
      const int d = n * 16 + (lane & 15);
      v16h vbf;
#pragma unroll
      for (int i = 0; i < 16; ++i)
        vbf[i] = (_Float16)Vb[(size_t)(kb + hi8 + i + (i & 8)) * HD + d];
      oacc[n] = __builtin_amdgcn_wmma_f32_16x16x32_f16(false, pa, false, vbf, (short)0,
                                                       oacc[n], false, false);
    }
  }

  // zero-fill strictly-above-diagonal weight region (k >= (qt+1)*16), coalesced NT float4s
  const int kz0 = (qt + 1) * 16;
  const v4f z4 = {0.f, 0.f, 0.f, 0.f};
  for (int r = 0; r < 16; ++r) {
    float* wr = Wb + (size_t)(q0 + r) * SEQ;
    for (int kk = kz0 + lane * 4; kk < SEQ; kk += 128)
      __builtin_nontemporal_store(z4, (v4f*)(wr + kk));
  }

  // store O: C-layout -> row q = v + hi8, col d = n*16 + lane&15 (64B coalesced per (n,v))
#pragma unroll
  for (int n = 0; n < 4; ++n)
#pragma unroll
    for (int v = 0; v < 8; ++v)
      Ob[(size_t)(q0 + v + hi8) * HD + n * 16 + (lane & 15)] = oacc[n][v];
}

extern "C" void kernel_launch(void* const* d_in, const int* in_sizes, int n_in,
                              void* d_out, int out_size, void* d_ws, size_t ws_size,
                              hipStream_t stream) {
  const float* q = (const float*)d_in[0];
  const float* k = (const float*)d_in[1];
  const float* v = (const float*)d_in[2];
  // d_in[3] = causal mask (derived analytically in-kernel)
  float* out = (float*)d_out;                       // [2,16,2048,64]
  float* w   = out + (size_t)2 * 16 * 2048 * 64;    // [2,16,2048,2048]
  dim3 grid(512), block(256);                        // 4096 waves: 32 bh x 128 q-tiles
  attn_fwd_causal<<<grid, block, 0, stream>>>(q, k, v, out, w);
}
